// GConv_7499012899094
// MI455X (gfx1250) — matrix operand order repaired and checked by reading, not compiled, not run on hardware
//
#include <hip/hip_runtime.h>

// GraphConv x5 + readout for MI455X (gfx1250, wave32, WMMA).
//
// Single bf16 operand matrix abig[N][768]:
//   cols   0..383 : bf16 copy of agg (agg accumulated in f32 via L2 atomics,
//                   then streamed to bf16 once per iteration)
//   cols 384..767 : x (bf16), written in place by the GEMM, gathered by the
//                   scatter kernel, read by the readout
// GEMM K-loop: one 24-step loop, pure {4x A b128 + 8x B b128 + 8x WMMA},
// no conversion VALU, no branches (rows are contiguous across agg|x halves).
// Weights pre-packed once into the exact WMMA B-fragment layout.

#define FDIM     384
#define ABIG_W   768                     // bf16 elements per abig row
#define KTILES   24                      // 768 / 32
#define CTILES   24                      // 384 / 16

#define B_NI_STRIDE  (32 * 8)            // uints between col-fragments
#define B_KT_STRIDE  (CTILES * 32 * 8)   // uints between k-fragments

typedef __attribute__((ext_vector_type(16))) __bf16        v16bf;
typedef __attribute__((ext_vector_type(8)))  float         v8f;
typedef __attribute__((ext_vector_type(8)))  unsigned int  v8u;

// ---- f32 -> bf16 (RNE) -----------------------------------------------------
static __device__ __forceinline__ unsigned int pk_bf16(float a, float b) {
  unsigned int ua = __float_as_uint(a);
  unsigned int ub = __float_as_uint(b);
  ua += 0x7FFFu + ((ua >> 16) & 1u);
  ub += 0x7FFFu + ((ub >> 16) & 1u);
  return (ua >> 16) | (ub & 0xFFFF0000u);
}

static __device__ __forceinline__ unsigned short bf16_1(float a) {
  unsigned int u = __float_as_uint(a);
  u += 0x7FFFu + ((u >> 16) & 1u);
  return (unsigned short)(u >> 16);
}

// ---------------------------------------------------------------------------
// Pack fused B = [W_rel^T ; W_root^T] (768 x 384) into WMMA B-fragment
// layout, bf16. Fragment (kt, ct): lane = 16*h + n holds column j = ct*16+n,
// K values kt*32 + 16*h + {0..15}, two bf16 per dword -> 8 dwords per lane.
// ---------------------------------------------------------------------------
__global__ void prep_B_kernel(const float* __restrict__ Wrel,
                              const float* __restrict__ Wroot,
                              unsigned int* __restrict__ Bpack) {
  int frag = blockIdx.x;                 // 0 .. KTILES*CTILES-1
  int kt = frag / CTILES;
  int ct = frag % CTILES;
  int lane = threadIdx.x;                // 0..31
  int h = lane >> 4, n = lane & 15;
  int j = ct * 16 + n;                   // output column
  unsigned int o[8];
#pragma unroll
  for (int v = 0; v < 8; ++v) {
    int k0 = kt * 32 + 16 * h + 2 * v;
    int k1 = k0 + 1;
    float f0 = (k0 < FDIM) ? Wrel[(size_t)j * FDIM + k0]
                           : Wroot[(size_t)j * FDIM + (k0 - FDIM)];
    float f1 = (k1 < FDIM) ? Wrel[(size_t)j * FDIM + k1]
                           : Wroot[(size_t)j * FDIM + (k1 - FDIM)];
    o[v] = pk_bf16(f0, f1);
  }
  uint4* q = (uint4*)(Bpack + (((size_t)frag * 32) + lane) * 8);
  q[0] = make_uint4(o[0], o[1], o[2], o[3]);
  q[1] = make_uint4(o[4], o[5], o[6], o[7]);
}

// ---------------------------------------------------------------------------
__global__ void zero_f32_kernel(float4* __restrict__ p, unsigned int n4) {
  unsigned int i = blockIdx.x * blockDim.x + threadIdx.x;
  if (i < n4) p[i] = make_float4(0.f, 0.f, 0.f, 0.f);
}

// Stream-convert an [N,384] f32 matrix into one half of abig (bf16).
// dstOff (uints): 0 = agg half, 192 = x half. One uint (2 bf16) per thread.
__global__ void cvt_half_kernel(const float* __restrict__ in,
                                unsigned int* __restrict__ abig_u32,
                                unsigned int n2, int dstOff) {
  unsigned int i = blockIdx.x * blockDim.x + threadIdx.x;
  if (i >= n2) return;
  unsigned int row = i / 192u;           // 192 f32-pairs per source row
  unsigned int c   = i % 192u;
  float2 f = ((const float2*)in)[i];
  abig_u32[(size_t)row * (ABIG_W / 2) + dstOff + c] = pk_bf16(f.x, f.y);
}

// ---------------------------------------------------------------------------
// agg[dst[e]] += x[src[e]]   — one block (96 threads) per edge.
// e = blockIdx.x is uniform -> src/dst are scalar loads; the 768B bf16 row
// gather (x half of abig) is fully coalesced; f32 atomics land in L2.
// ---------------------------------------------------------------------------
__global__ void scatter_add_kernel(const unsigned short* __restrict__ abig,
                                   const int* __restrict__ src,
                                   const int* __restrict__ dst,
                                   float* __restrict__ agg) {
  const int e = blockIdx.x;
  const int c = threadIdx.x;                 // 0..95 (4 bf16 each)
  const int s = src[e];
  const int d = dst[e];
  uint2 v = ((const uint2*)(abig + (size_t)s * ABIG_W + FDIM))[c];
  float* ap = agg + (size_t)d * FDIM + (size_t)c * 4u;
  atomicAdd(ap + 0, __uint_as_float((v.x & 0xFFFFu) << 16));
  atomicAdd(ap + 1, __uint_as_float(v.x & 0xFFFF0000u));
  atomicAdd(ap + 2, __uint_as_float((v.y & 0xFFFFu) << 16));
  atomicAdd(ap + 3, __uint_as_float(v.y & 0xFFFF0000u));
}

// ---------------------------------------------------------------------------
// Fragment loaders (pure loads, no conversion)
// ---------------------------------------------------------------------------
static __device__ __forceinline__ v16bf load_a_frag(const unsigned short* p) {
  // p = &A[row][k0 + 16h]; lane 16h+m: K runs {16h..16h+15}
  uint4 b0 = *(const uint4*)p;
  uint4 b1 = *(const uint4*)(p + 8);
  v8u u;
  u[0] = b0.x; u[1] = b0.y; u[2] = b0.z; u[3] = b0.w;
  u[4] = b1.x; u[5] = b1.y; u[6] = b1.z; u[7] = b1.w;
  return __builtin_bit_cast(v16bf, u);
}

static __device__ __forceinline__ v16bf load_b_frag(const unsigned int* __restrict__ q) {
  uint4 b0 = ((const uint4*)q)[0];
  uint4 b1 = ((const uint4*)q)[1];
  v8u u;
  u[0] = b0.x; u[1] = b0.y; u[2] = b0.z; u[3] = b0.w;
  u[4] = b1.x; u[5] = b1.y; u[6] = b1.z; u[7] = b1.w;
  return __builtin_bit_cast(v16bf, u);
}

// ---------------------------------------------------------------------------
// Fused GEMM: x half of abig = relu(abig @ Bpack). Row-local => in-place.
// Block = 256 threads (8 waves), tile 128 rows x 128 cols.
// Wave grid 4(M) x 2(N): each wave owns 32 rows x 64 cols = 2x4 WMMA tiles.
// Tail rows: clamp A row addresses (loop-invariant), guard stores.
// ---------------------------------------------------------------------------
__global__ __launch_bounds__(256) void gemm_fused_kernel(
    unsigned short* abig, const unsigned int* __restrict__ Bpack, int Nn) {
  const int wid  = threadIdx.x >> 5;
  const int lane = threadIdx.x & 31;
  const int h  = lane >> 4;          // half-wave
  const int ln = lane & 15;          // lane-in-half
  const int waveM = wid & 3;
  const int waveN = wid >> 2;
  const int rowBase = blockIdx.x * 128 + waveM * 32;   // two 16-row groups
  const int colBase = blockIdx.y * 128 + waveN * 64;   // four 16-col tiles
  const int ct0 = colBase >> 4;

  int r0 = rowBase + ln;          r0 = (r0 < Nn) ? r0 : (Nn - 1);
  int r1 = rowBase + 16 + ln;     r1 = (r1 < Nn) ? r1 : (Nn - 1);
  const unsigned short* a0 = abig + (size_t)r0 * ABIG_W + 16 * h;
  const unsigned short* a1 = abig + (size_t)r1 * ABIG_W + 16 * h;

  v8f acc[2][4];
#pragma unroll
  for (int mi = 0; mi < 2; ++mi)
#pragma unroll
    for (int ni = 0; ni < 4; ++ni) { v8f z = {}; acc[mi][ni] = z; }

  const unsigned int* bptr = Bpack + ((size_t)ct0 * 32 + lane) * 8;

  // ---- single K loop over the contiguous [agg | x] row (768 bf16)
  for (int kt = 0; kt < KTILES; ++kt) {
    __builtin_prefetch(a0 + 64, 0, 1);     // 2 K-steps ahead
    __builtin_prefetch(a1 + 64, 0, 1);
    v16bf af0 = load_a_frag(a0);
    v16bf af1 = load_a_frag(a1);
    v16bf bf0 = load_b_frag(bptr);
    v16bf bf1 = load_b_frag(bptr + 1 * B_NI_STRIDE);
    v16bf bf2 = load_b_frag(bptr + 2 * B_NI_STRIDE);
    v16bf bf3 = load_b_frag(bptr + 3 * B_NI_STRIDE);
    acc[0][0] = __builtin_amdgcn_wmma_f32_16x16x32_bf16(false, af0, false, bf0,
                                                        (short)0, acc[0][0], false, false);
    acc[0][1] = __builtin_amdgcn_wmma_f32_16x16x32_bf16(false, af0, false, bf1,
                                                        (short)0, acc[0][1], false, false);
    acc[0][2] = __builtin_amdgcn_wmma_f32_16x16x32_bf16(false, af0, false, bf2,
                                                        (short)0, acc[0][2], false, false);
    acc[0][3] = __builtin_amdgcn_wmma_f32_16x16x32_bf16(false, af0, false, bf3,
                                                        (short)0, acc[0][3], false, false);
    acc[1][0] = __builtin_amdgcn_wmma_f32_16x16x32_bf16(false, af1, false, bf0,
                                                        (short)0, acc[1][0], false, false);
    acc[1][1] = __builtin_amdgcn_wmma_f32_16x16x32_bf16(false, af1, false, bf1,
                                                        (short)0, acc[1][1], false, false);
    acc[1][2] = __builtin_amdgcn_wmma_f32_16x16x32_bf16(false, af1, false, bf2,
                                                        (short)0, acc[1][2], false, false);
    acc[1][3] = __builtin_amdgcn_wmma_f32_16x16x32_bf16(false, af1, false, bf3,
                                                        (short)0, acc[1][3], false, false);
    a0 += 32; a1 += 32; bptr += B_KT_STRIDE;
  }

  // ---- store relu'd bf16 into x half (C/D: lane 16h+n -> (M=v+8h, N=n))
#pragma unroll
  for (int mi = 0; mi < 2; ++mi) {
#pragma unroll
    for (int ni = 0; ni < 4; ++ni) {
      int col = colBase + ni * 16 + ln;
#pragma unroll
      for (int v = 0; v < 8; ++v) {
        int row = rowBase + mi * 16 + v + 8 * h;
        if (row < Nn)
          abig[(size_t)row * ABIG_W + FDIM + col] =
              bf16_1(fmaxf(acc[mi][ni][v], 0.f));
      }
    }
  }
}

// ---------------------------------------------------------------------------
// out[n] = x[n,:] . wlin   — one wave per node, wave32 xor-reduction.
// ---------------------------------------------------------------------------
__global__ void readout_kernel(const unsigned short* __restrict__ abig,
                               const float* __restrict__ wlin,
                               float* __restrict__ out, int Nn) {
  int wid = threadIdx.x >> 5, lane = threadIdx.x & 31;
  int node = blockIdx.x * 8 + wid;
  if (node >= Nn) return;
  const unsigned int* xr =
      (const unsigned int*)(abig + (size_t)node * ABIG_W + FDIM);
  float s = 0.f;
#pragma unroll
  for (int j = 0; j < FDIM / 64; ++j) {      // 6 uints (12 bf16) per lane
    unsigned int idx = lane + j * 32;
    unsigned int v = xr[idx];
    s += __uint_as_float((v & 0xFFFFu) << 16) * wlin[2 * idx];
    s += __uint_as_float(v & 0xFFFF0000u)     * wlin[2 * idx + 1];
  }
#pragma unroll
  for (int off = 16; off > 0; off >>= 1)
    s += __shfl_xor(s, off, 32);
  if (lane == 0) out[node] = s;
}

// ---------------------------------------------------------------------------
extern "C" void kernel_launch(void* const* d_in, const int* in_sizes, int n_in,
                              void* d_out, int out_size, void* d_ws, size_t ws_size,
                              hipStream_t stream) {
  const float* x_in  = (const float*)d_in[0];
  const int*   ei    = (const int*)d_in[1];
  // d_in[2] = batch (unused by forward)
  const float* Wrel  = (const float*)d_in[3];
  const float* Wroot = (const float*)d_in[4];
  const float* Wlin  = (const float*)d_in[5];
  float* out = (float*)d_out;

  const int Nn = in_sizes[0] / FDIM;     // 50000
  const int Ee = in_sizes[1] / 2;        // 1600000
  const int* src = ei;
  const int* dst = ei + Ee;

  // workspace: abig (N*768 bf16) | agg (N*384 f32) | Bpack (768*384 bf16)
  char* ws = (char*)d_ws;
  const size_t abig_bytes = (size_t)Nn * ABIG_W * sizeof(unsigned short);
  unsigned short* abig = (unsigned short*)ws;
  float* agg           = (float*)(ws + abig_bytes);
  unsigned int* Bpack  =
      (unsigned int*)(ws + abig_bytes + (size_t)Nn * FDIM * sizeof(float));

  prep_B_kernel<<<KTILES * CTILES, 32, 0, stream>>>(Wrel, Wroot, Bpack);

  const unsigned int n2 = (unsigned int)((size_t)Nn * FDIM / 2);
  const unsigned int n4 = (unsigned int)((size_t)Nn * FDIM / 4);

  // x_in -> x half of abig (bf16)
  cvt_half_kernel<<<(n2 + 255) / 256, 256, 0, stream>>>(
      x_in, (unsigned int*)abig, n2, 192);

  dim3 ggrid((Nn + 127) / 128, FDIM / 128);

  for (int it = 0; it < 5; ++it) {
    zero_f32_kernel<<<(n4 + 255) / 256, 256, 0, stream>>>((float4*)agg, n4);
    scatter_add_kernel<<<Ee, 96, 0, stream>>>(abig, src, dst, agg);
    // agg (f32) -> agg half of abig (bf16), streaming pass (L2-resident)
    cvt_half_kernel<<<(n2 + 255) / 256, 256, 0, stream>>>(
        agg, (unsigned int*)abig, n2, 0);
    // row-local GEMM, in-place into the x half of abig
    gemm_fused_kernel<<<ggrid, 256, 0, stream>>>(abig, Bpack, Nn);
  }

  readout_kernel<<<(Nn + 7) / 8, 256, 0, stream>>>(abig, Wlin, out, Nn);
}
